// EGNNLiteLayer_12816182411328
// MI455X (gfx1250) — matrix operand order repaired and checked by reading, not compiled, not run on hardware
//
#include <hip/hip_runtime.h>
#include <hip/hip_bf16.h>

typedef _Float16 hv4  __attribute__((ext_vector_type(4)));
typedef _Float16 hv8  __attribute__((ext_vector_type(8)));
typedef _Float16 hv16 __attribute__((ext_vector_type(16)));
typedef float    fv8  __attribute__((ext_vector_type(8)));

#define D_MODEL 128
#define D_EDGE  64
#define KE      288   // 266 padded to 9*32
#define KN      192   // node-MLP K (128+64)
#define HID     256   // node-MLP hidden

__device__ __forceinline__ float silu_f(float x)    { return x / (1.f + __expf(-x)); }
__device__ __forceinline__ float sigmoid_f(float x) { return 1.f / (1.f + __expf(-x)); }

// A-fragment for v_wmma_f32_16x16x32_f16 from an LDS row (row m = lane&15).
// lane<16: halves = K k0+0..15 ; lane>=16: K k0+16..31.
__device__ __forceinline__ hv16 load_a_lds(const _Float16* rowp, int kk, int hi) {
  const _Float16* ap = rowp + kk + 16 * hi;
  hv8 lo = *(const hv8*)(ap);
  hv8 hh = *(const hv8*)(ap + 8);
  return __builtin_shufflevector(lo, hh, 0,1,2,3,4,5,6,7,8,9,10,11,12,13,14,15);
}

__global__ void zero_f32(float* p, long long n) {
  long long i = (long long)blockIdx.x * blockDim.x + threadIdx.x;
  long long s = (long long)gridDim.x * blockDim.x;
  for (; i < n; i += s) p[i] = 0.f;
}

// dst[n*Kpad + k] = (k<K) ? src[k*Nn + n] : 0   (f32 -> f16, K-major for B frags)
__global__ void prep_transpose_f16(const float* __restrict__ src, _Float16* __restrict__ dst,
                                   int K, int Nn, int Kpad) {
  int idx = blockIdx.x * blockDim.x + threadIdx.x;
  int total = Nn * Kpad;
  if (idx >= total) return;
  int n = idx / Kpad, k = idx - n * Kpad;
  dst[idx] = (k < K) ? (_Float16)src[(size_t)k * Nn + n] : (_Float16)0.f;
}

// ---------------- Edge kernel: 4 waves/block, 16 edges per wave ----------------
__global__ __launch_bounds__(128) void egnn_edge_kernel(
    const float* __restrict__ H, const float* __restrict__ xyz,
    const long long* __restrict__ eidx, const float* __restrict__ estruct,
    const float* __restrict__ rest,
    const _Float16* __restrict__ We1t, const float* __restrict__ be1,
    const _Float16* __restrict__ We2t, const float* __restrict__ be2,
    const float* __restrict__ Wg1, const float* __restrict__ bg1,
    const float* __restrict__ Wg2, const float* __restrict__ bg2,
    float* __restrict__ agg, int E)
{
  __shared__ __align__(32) _Float16 Ein[4][16][KE];      // 36.9 KB
  __shared__ __align__(32) _Float16 Act[4][16][D_EDGE];  // 8.2 KB
  __shared__ int   IdxI[4][16];
  __shared__ int   IdxJ[4][16];
  __shared__ float Gv[4][16];

  const int tid   = threadIdx.x;
  const int wave  = tid >> 5;
  const int lane  = tid & 31;
  const int lrow  = lane & 15;
  const int hi    = lane >> 4;
  const int ebase = blockIdx.x * 64 + wave * 16;

  // ---- stage 1: indices, geometric features, gate MLP (lanes 0..15) ----
  if (hi == 0) {
    const int  m     = lrow;
    const int  e     = ebase + m;
    const bool valid = (e < E);
    const int  ec    = valid ? e : (E - 1);
    const long long ii = eidx[ec];
    const long long jj = eidx[(size_t)E + ec];
    IdxI[wave][m] = (int)ii;
    IdxJ[wave][m] = (int)jj;
    float dx = xyz[ii*3+0] - xyz[jj*3+0];
    float dy = xyz[ii*3+1] - xyz[jj*3+1];
    float dz = xyz[ii*3+2] - xyz[jj*3+2];
    float dist2 = dx*dx + dy*dy + dz*dz;
    float dist  = sqrtf(dist2 + 1e-9f);
    float rl    = rest[ec];
    float delta = (dist - rl) / (rl + 1e-9f);
    float st[8];
    #pragma unroll
    for (int k = 0; k < 8; ++k) st[k] = estruct[(size_t)ec*8 + k];
    Ein[wave][m][256] = (_Float16)dist2;
    Ein[wave][m][257] = (_Float16)delta;
    #pragma unroll
    for (int k = 0; k < 8; ++k) Ein[wave][m][258+k] = (_Float16)st[k];
    #pragma unroll
    for (int k = 266; k < KE; ++k) Ein[wave][m][k] = (_Float16)0.f;
    // gate: 10 -> 32 (silu) -> 1 (sigmoid); tiny, VALU, weights L2-resident
    float gacc = bg2[0];
    for (int h = 0; h < 32; ++h) {
      float t = bg1[h] + dist2 * Wg1[h] + delta * Wg1[32 + h];
      #pragma unroll
      for (int k = 0; k < 8; ++k) t += st[k] * Wg1[(2 + k)*32 + h];
      gacc += silu_f(t) * Wg2[h];
    }
    Gv[wave][m] = valid ? sigmoid_f(gacc) : 0.f;   // invalid edges contribute 0
  }
  __syncthreads();

  // ---- stage 2: gather H[i], H[j] rows into Ein (f32 -> f16), 2 lanes/edge ----
  {
    const int m   = lane >> 1;
    const int sub = lane & 1;
    const long long ni = IdxI[wave][m];
    const long long nj = IdxJ[wave][m];
    const float4* hri = (const float4*)(H + (size_t)ni * D_MODEL);
    const float4* hrj = (const float4*)(H + (size_t)nj * D_MODEL);
    _Float16* di = &Ein[wave][m][sub * 64];
    _Float16* dj = &Ein[wave][m][128 + sub * 64];
    #pragma unroll
    for (int v = 0; v < 16; ++v) {
      float4 f = hri[sub*16 + v];
      hv4 h = { (_Float16)f.x, (_Float16)f.y, (_Float16)f.z, (_Float16)f.w };
      *(hv4*)(di + v*4) = h;
    }
    #pragma unroll
    for (int v = 0; v < 16; ++v) {
      float4 f = hrj[sub*16 + v];
      hv4 h = { (_Float16)f.x, (_Float16)f.y, (_Float16)f.z, (_Float16)f.w };
      *(hv4*)(dj + v*4) = h;
    }
  }
  __syncthreads();

  // ---- GEMM1: [16 x 288] @ [288 x 64], 9 K-steps x 4 N-tiles of WMMA ----
  fv8 acc[4] = {};
  const _Float16* arow = &Ein[wave][lrow][0];
  #pragma unroll
  for (int kk = 0; kk < KE; kk += 32) {
    hv16 a = load_a_lds(arow, kk, hi);
    #pragma unroll
    for (int t = 0; t < 4; ++t) {
      const _Float16* bp = We1t + (size_t)(t*16 + lrow) * KE + kk + 16*hi;
      hv16 b = *(const hv16*)bp;
      acc[t] = __builtin_amdgcn_wmma_f32_16x16x32_f16(false, a, false, b, (short)0, acc[t], false, false);
    }
  }
  #pragma unroll
  for (int t = 0; t < 4; ++t) {
    const int n = t*16 + lrow;
    const float bias = be1[n];
    #pragma unroll
    for (int r = 0; r < 8; ++r)
      Act[wave][r + 8*hi][n] = (_Float16)silu_f(acc[t][r] + bias);
  }
  __syncthreads();

  // ---- GEMM2: [16 x 64] @ [64 x 64], then gate + atomic scatter-add ----
  fv8 acc2[4] = {};
  const _Float16* arow2 = &Act[wave][lrow][0];
  #pragma unroll
  for (int kk = 0; kk < D_EDGE; kk += 32) {
    hv16 a = load_a_lds(arow2, kk, hi);
    #pragma unroll
    for (int t = 0; t < 4; ++t) {
      const _Float16* bp = We2t + (size_t)(t*16 + lrow) * D_EDGE + kk + 16*hi;
      hv16 b = *(const hv16*)bp;
      acc2[t] = __builtin_amdgcn_wmma_f32_16x16x32_f16(false, a, false, b, (short)0, acc2[t], false, false);
    }
  }
  #pragma unroll
  for (int t = 0; t < 4; ++t) {
    const int n = t*16 + lrow;
    const float bias = be2[n];
    #pragma unroll
    for (int r = 0; r < 8; ++r) {
      const int m = r + 8*hi;
      float v = silu_f(acc2[t][r] + bias) * Gv[wave][m];
      atomicAdd(agg + (size_t)IdxI[wave][m] * D_EDGE + n, v);  // L2-resident agg
    }
  }
}

// ---------------- Node kernel: 4 waves/block, 16 nodes per wave ----------------
__global__ __launch_bounds__(128) void egnn_node_kernel(
    const float* __restrict__ H, const float* __restrict__ agg,
    const _Float16* __restrict__ Wn1t, const float* __restrict__ bn1,
    const _Float16* __restrict__ Wn2t, const float* __restrict__ bn2,
    const float* __restrict__ lng, const float* __restrict__ lnb,
    float* __restrict__ out, int N)
{
  __shared__ __align__(32) _Float16 Nin[4][16][KN];   // 24.6 KB
  __shared__ __align__(32) _Float16 Hid[4][16][HID];  // 32.8 KB

  const int tid   = threadIdx.x;
  const int wave  = tid >> 5;
  const int lane  = tid & 31;
  const int lrow  = lane & 15;
  const int hi    = lane >> 4;
  const int nbase = blockIdx.x * 64 + wave * 16;

  // ---- fill Nin = [H(128) | agg(64)] as f16, 2 lanes/node ----
  {
    const int m   = lane >> 1;
    const int sub = lane & 1;
    int node = nbase + m; if (node >= N) node = N - 1;
    const float4* hr = (const float4*)(H   + (size_t)node * D_MODEL);
    const float4* ar = (const float4*)(agg + (size_t)node * D_EDGE);
    _Float16* dh = &Nin[wave][m][sub * 64];
    #pragma unroll
    for (int v = 0; v < 16; ++v) {
      float4 f = hr[sub*16 + v];
      hv4 h = { (_Float16)f.x, (_Float16)f.y, (_Float16)f.z, (_Float16)f.w };
      *(hv4*)(dh + v*4) = h;
    }
    _Float16* da = &Nin[wave][m][128 + sub * 32];
    #pragma unroll
    for (int v = 0; v < 8; ++v) {
      float4 f = ar[sub*8 + v];
      hv4 h = { (_Float16)f.x, (_Float16)f.y, (_Float16)f.z, (_Float16)f.w };
      *(hv4*)(da + v*4) = h;
    }
  }
  __syncthreads();

  // ---- GEMM1: [16 x 192] @ [192 x 256] (silu), chunked over N in 4 x 64 ----
  const _Float16* arow = &Nin[wave][lrow][0];
  #pragma unroll
  for (int nc = 0; nc < 4; ++nc) {
    fv8 acc[4] = {};
    #pragma unroll
    for (int kk = 0; kk < KN; kk += 32) {
      hv16 a = load_a_lds(arow, kk, hi);
      #pragma unroll
      for (int t = 0; t < 4; ++t) {
        const int n = nc*64 + t*16 + lrow;
        const _Float16* bp = Wn1t + (size_t)n * KN + kk + 16*hi;
        hv16 b = *(const hv16*)bp;
        acc[t] = __builtin_amdgcn_wmma_f32_16x16x32_f16(false, a, false, b, (short)0, acc[t], false, false);
      }
    }
    #pragma unroll
    for (int t = 0; t < 4; ++t) {
      const int n = nc*64 + t*16 + lrow;
      const float bias = bn1[n];
      #pragma unroll
      for (int r = 0; r < 8; ++r)
        Hid[wave][r + 8*hi][n] = (_Float16)silu_f(acc[t][r] + bias);
    }
  }
  __syncthreads();

  // ---- GEMM2: [16 x 256] @ [256 x 128] ----
  fv8 acc2[8] = {};
  const _Float16* arow2 = &Hid[wave][lrow][0];
  #pragma unroll
  for (int kk = 0; kk < HID; kk += 32) {
    hv16 a = load_a_lds(arow2, kk, hi);
    #pragma unroll
    for (int t = 0; t < 8; ++t) {
      const _Float16* bp = Wn2t + (size_t)(t*16 + lrow) * HID + kk + 16*hi;
      hv16 b = *(const hv16*)bp;
      acc2[t] = __builtin_amdgcn_wmma_f32_16x16x32_f16(false, a, false, b, (short)0, acc2[t], false, false);
    }
  }

  // ---- residual + LayerNorm (cross-lane reduce over the 16 lanes of a row) ----
  #pragma unroll
  for (int r = 0; r < 8; ++r) {
    const int m    = r + 8*hi;
    const int node = nbase + m;
    const int ncl  = (node < N) ? node : (N - 1);
    float s = 0.f;
    #pragma unroll
    for (int t = 0; t < 8; ++t) {
      const int n = t*16 + lrow;
      float v = acc2[t][r] + bn2[n] + H[(size_t)ncl * D_MODEL + n];
      acc2[t][r] = v;
      s += v;
    }
    s += __shfl_xor(s, 1, 32); s += __shfl_xor(s, 2, 32);
    s += __shfl_xor(s, 4, 32); s += __shfl_xor(s, 8, 32);
    const float mu = s * (1.f / 128.f);
    float q = 0.f;
    #pragma unroll
    for (int t = 0; t < 8; ++t) { float d = acc2[t][r] - mu; q += d * d; }
    q += __shfl_xor(q, 1, 32); q += __shfl_xor(q, 2, 32);
    q += __shfl_xor(q, 4, 32); q += __shfl_xor(q, 8, 32);
    const float inv = rsqrtf(q * (1.f / 128.f) + 1e-5f);
    if (node < N) {
      #pragma unroll
      for (int t = 0; t < 8; ++t) {
        const int n = t*16 + lrow;
        out[(size_t)node * D_MODEL + n] = (acc2[t][r] - mu) * inv * lng[n] + lnb[n];
      }
    }
  }
}

extern "C" void kernel_launch(void* const* d_in, const int* in_sizes, int n_in,
                              void* d_out, int out_size, void* d_ws, size_t ws_size,
                              hipStream_t stream)
{
  const float*     H       = (const float*)d_in[0];
  const float*     xyz     = (const float*)d_in[1];
  const long long* eidx    = (const long long*)d_in[2];   // int64
  const float*     estruct = (const float*)d_in[3];
  const float*     rest    = (const float*)d_in[4];
  const float*     We1     = (const float*)d_in[5];
  const float*     be1     = (const float*)d_in[6];
  const float*     We2     = (const float*)d_in[7];
  const float*     be2     = (const float*)d_in[8];
  const float*     Wg1     = (const float*)d_in[9];
  const float*     bg1     = (const float*)d_in[10];
  const float*     Wg2     = (const float*)d_in[11];
  const float*     bg2     = (const float*)d_in[12];
  const float*     Wn1     = (const float*)d_in[13];
  const float*     bn1     = (const float*)d_in[14];
  const float*     Wn2     = (const float*)d_in[15];
  const float*     bn2     = (const float*)d_in[16];
  const float*     lng     = (const float*)d_in[17];
  const float*     lnb     = (const float*)d_in[18];

  const int N = in_sizes[0] / D_MODEL;   // B == 1
  const int E = in_sizes[4];

  // workspace layout: agg (f32 N x 64) | f16 transposed weights (256B aligned)
  char*  ws  = (char*)d_ws;
  size_t off = 0;
  float* agg = (float*)(ws + off); off += (size_t)N * D_EDGE * sizeof(float);
  off = (off + 255) & ~(size_t)255;
  _Float16* We1t = (_Float16*)(ws + off); off += (size_t)64  * KE  * 2;
  off = (off + 255) & ~(size_t)255;
  _Float16* We2t = (_Float16*)(ws + off); off += (size_t)64  * 64  * 2;
  off = (off + 255) & ~(size_t)255;
  _Float16* Wn1t = (_Float16*)(ws + off); off += (size_t)256 * KN  * 2;
  off = (off + 255) & ~(size_t)255;
  _Float16* Wn2t = (_Float16*)(ws + off); off += (size_t)128 * HID * 2;

  zero_f32<<<512, 256, 0, stream>>>(agg, (long long)N * D_EDGE);
  prep_transpose_f16<<<(64*KE   + 255)/256, 256, 0, stream>>>(We1, We1t, 266, 64,  KE);
  prep_transpose_f16<<<(64*64   + 255)/256, 256, 0, stream>>>(We2, We2t, 64,  64,  64);
  prep_transpose_f16<<<(256*KN  + 255)/256, 256, 0, stream>>>(Wn1, Wn1t, KN,  256, KN);
  prep_transpose_f16<<<(128*HID + 255)/256, 256, 0, stream>>>(Wn2, Wn2t, HID, 128, HID);

  egnn_edge_kernel<<<(E + 63)/64, 128, 0, stream>>>(H, xyz, eidx, estruct, rest,
                                                    We1t, be1, We2t, be2,
                                                    Wg1, bg1, Wg2, bg2, agg, E);
  egnn_node_kernel<<<(N + 63)/64, 128, 0, stream>>>(H, agg, Wn1t, bn1, Wn2t, bn2,
                                                    lng, lnb, (float*)d_out, N);
}
